// ClusteringLayer_36601711296729
// MI455X (gfx1250) — matrix operand order repaired and checked by reading, not compiled, not run on hardware
//
#include <hip/hip_runtime.h>
#include <stdint.h>

typedef _Float16 half8  __attribute__((ext_vector_type(8)));
typedef _Float16 half16 __attribute__((ext_vector_type(16)));
typedef float    float8 __attribute__((ext_vector_type(8)));

#define N_PTS 131072
#define K_CL  1024
#define FDIM  256
#define K_TILE 128        // clusters per workgroup
#define ROWS_PER_BLK 128  // points staged per inner block
#define NBLK 8            // point-blocks per workgroup (1024 points/WG)
#define PITCH 264         // f16 elems per LDS row (256 + 8 pad -> conflict-free b128)
#define THREADS 256       // 8 waves (wave32)

// Dynamic LDS layout (bytes)
#define C_OFF   0
#define C_BYTES (K_TILE * PITCH * 2)            // 67584
#define X_OFF   (C_OFF + C_BYTES)
#define X_BYTES (ROWS_PER_BLK * PITCH * 2)      // 67584
#define C2_OFF  (X_OFF + X_BYTES)
#define X2_OFF  (C2_OFF + K_TILE * 4)
#define MK_OFF  (X2_OFF + ROWS_PER_BLK * 4)
#define SMEM_BYTES (MK_OFF + K_TILE * 8)        // ~137 KB (WGP has 320 KB)

static __device__ __forceinline__ unsigned long long umin64(unsigned long long a,
                                                            unsigned long long b) {
  return a < b ? a : b;
}

__global__ void init_keys_kernel(unsigned long long* __restrict__ gkey) {
  int i = blockIdx.x * blockDim.x + threadIdx.x;
  if (i < K_CL) gkey[i] = ~0ull;
}

__global__ __launch_bounds__(THREADS)
void argmin_dist_kernel(const float* __restrict__ x, const float* __restrict__ c,
                        unsigned long long* __restrict__ gkey) {
  extern __shared__ char smem[];
  _Float16* c_lds = (_Float16*)(smem + C_OFF);
  _Float16* x_lds = (_Float16*)(smem + X_OFF);
  float*    c2    = (float*)(smem + C2_OFF);
  float*    x2    = (float*)(smem + X2_OFF);
  unsigned long long* mk = (unsigned long long*)(smem + MK_OFF);

  const int tid  = threadIdx.x;
  const int lane = tid & 31;
  const int wave = tid >> 5;
  const int kbase = blockIdx.x * K_TILE;

  if (tid < K_TILE) mk[tid] = ~0ull;

  // ---- Stage cluster tile once: K_TILE x FDIM f32 -> f16 in LDS ----
  for (int i = tid; i < K_TILE * (FDIM / 4); i += THREADS) {
    const int row = i >> 6;         // FDIM/4 == 64 float4 per row
    const int c4  = i & 63;
    const float4 v = ((const float4*)c)[(size_t)(kbase + row) * (FDIM / 4) + c4];
    _Float16* dst = &c_lds[row * PITCH + c4 * 4];
    dst[0] = (_Float16)v.x; dst[1] = (_Float16)v.y;
    dst[2] = (_Float16)v.z; dst[3] = (_Float16)v.w;
  }
  __syncthreads();
  if (tid < K_TILE) {
    float s = 0.f;
    for (int f = 0; f < FDIM; ++f) { float t = (float)c_lds[tid * PITCH + f]; s += t * t; }
    c2[tid] = s;
  }
  __syncthreads();

  const int rbase = wave * 16;                 // this wave's 16 point-rows
  const int arow  = rbase + (lane & 15);
  const int akoff = (lane >> 4) * 8;           // A: K 0-7 / 8-15 per lane half
  const int bkoff = (lane >> 4) * 8;           // B: K 0-7 / 8-15 per lane half
  const int mrow_off = (lane >> 4) * 8;        // C/D: M = r (lanes 0-15), r+8 (16-31)

  for (int nb = 0; nb < NBLK; ++nb) {
    const int nbase = (blockIdx.y * NBLK + nb) * ROWS_PER_BLK;
    __syncthreads();  // previous block's x_lds / x2 consumers done
    for (int i = tid; i < ROWS_PER_BLK * (FDIM / 4); i += THREADS) {
      const int row = i >> 6;
      const int c4  = i & 63;
      const float4 v = ((const float4*)x)[(size_t)(nbase + row) * (FDIM / 4) + c4];
      _Float16* dst = &x_lds[row * PITCH + c4 * 4];
      dst[0] = (_Float16)v.x; dst[1] = (_Float16)v.y;
      dst[2] = (_Float16)v.z; dst[3] = (_Float16)v.w;
    }
    __syncthreads();
    if (tid < ROWS_PER_BLK) {
      float s = 0.f;
      for (int f = 0; f < FDIM; ++f) { float t = (float)x_lds[tid * PITCH + f]; s += t * t; }
      x2[tid] = s;
    }
    __syncthreads();

    // ---- 16 x 128 strip per wave: 8 accumulators, 8 F-chunks of 32 ----
    float8 acc[8] = {};
#pragma unroll
    for (int fc = 0; fc < 8; ++fc) {
      const int fbase = fc * 32;
      union { half16 v; half8 h[2]; } A;
      A.h[0] = *(const half8*)&x_lds[arow * PITCH + fbase + akoff];
      A.h[1] = *(const half8*)&x_lds[arow * PITCH + fbase + akoff + 16];
#pragma unroll
      for (int ct = 0; ct < 8; ++ct) {
        const int crow = ct * 16 + (lane & 15);
        union { half16 v; half8 h[2]; } B;
        B.h[0] = *(const half8*)&c_lds[crow * PITCH + fbase + bkoff];
        B.h[1] = *(const half8*)&c_lds[crow * PITCH + fbase + bkoff + 16];
        acc[ct] = __builtin_amdgcn_wmma_f32_16x16x32_f16(
            false, A.v, false, B.v, (short)0, acc[ct], false, false);
      }
    }

    // ---- per-column argmin: local 8 rows -> lane^16 merge -> LDS u64 min ----
#pragma unroll
    for (int ct = 0; ct < 8; ++ct) {
      unsigned long long best = ~0ull;
#pragma unroll
      for (int r = 0; r < 8; ++r) {
        const int m = r + mrow_off;  // tile row 0..15
        float d2 = x2[rbase + m] + c2[ct * 16 + (lane & 15)] - 2.0f * acc[ct][r];
        d2 = fmaxf(d2, 0.0f);        // monotone u32 key for d2 >= 0
        const unsigned long long key =
            ((unsigned long long)__float_as_uint(d2) << 32) |
            (unsigned)(nbase + rbase + m);
        best = umin64(best, key);
      }
      const unsigned long long other =
          (unsigned long long)__shfl_xor((long long)best, 16, 32);
      best = umin64(best, other);
      if (lane < 16) atomicMin(&mk[ct * 16 + (lane & 15)], best);
    }
  }

  __syncthreads();
  if (tid < K_TILE) atomicMin(&gkey[kbase + tid], mk[tid]);
}

__global__ void gather_kernel(const float* __restrict__ x,
                              const unsigned long long* __restrict__ gkey,
                              float* __restrict__ out) {
  const int k = blockIdx.x;
  const unsigned idx = (unsigned)(gkey[k] & 0xffffffffull);
  const float4* src = (const float4*)(x + (size_t)idx * FDIM);
  float4* dst = (float4*)(out + (size_t)k * FDIM);
  dst[threadIdx.x] = src[threadIdx.x];  // 64 threads x float4 = 256 floats
}

extern "C" void kernel_launch(void* const* d_in, const int* in_sizes, int n_in,
                              void* d_out, int out_size, void* d_ws, size_t ws_size,
                              hipStream_t stream) {
  (void)in_sizes; (void)n_in; (void)out_size; (void)ws_size;
  const float* x = (const float*)d_in[0];
  const float* c = (const float*)d_in[1];
  float* out = (float*)d_out;
  unsigned long long* gkey = (unsigned long long*)d_ws;  // K_CL u64 keys

  init_keys_kernel<<<dim3((K_CL + 255) / 256), dim3(256), 0, stream>>>(gkey);

  dim3 grid(K_CL / K_TILE, N_PTS / (ROWS_PER_BLK * NBLK));  // (8, 128)
  argmin_dist_kernel<<<grid, dim3(THREADS), SMEM_BYTES, stream>>>(x, c, gkey);

  gather_kernel<<<dim3(K_CL), dim3(FDIM / 4), 0, stream>>>(x, gkey, out);
}
